// Model_24412594111261
// MI455X (gfx1250) — matrix operand order repaired and checked by reading, not compiled, not run on hardware
//
#include <hip/hip_runtime.h>

typedef _Float16 h16;
typedef __attribute__((ext_vector_type(16))) _Float16 v16h;
typedef __attribute__((ext_vector_type(8)))  _Float16 v8h;
typedef __attribute__((ext_vector_type(8)))  float    v8f;

#define NN 50000
#define NE 800000

static __device__ __forceinline__ v16h mk16(v8h lo, v8h hi) {
  return __builtin_shufflevector(lo, hi, 0,1,2,3,4,5,6,7,8,9,10,11,12,13,14,15);
}
static __device__ __forceinline__ v8f wmma32(v16h a, v16h b, v8f c) {
  return __builtin_amdgcn_wmma_f32_16x16x32_f16(false, a, false, b, (short)0, c, false, false);
}
// B fragment: lane holds col nt*16+(lane&15); elems0-7 @ kt*32+(lane>=16?8:0), elems8-15 @ +16
static __device__ __forceinline__ v16h load_b(const h16* __restrict__ W, int K, int kt, int nt, int lane) {
  int wrow = nt * 16 + (lane & 15);
  const h16* p = W + (size_t)wrow * K + kt * 32 + ((lane >> 4) << 3);
  return mk16(*(const v8h*)p, *(const v8h*)(p + 16));
}
// A fragment from a contiguous fp16 row: elems0-7 @ kbase+(lane>=16?8:0), elems8-15 @ +16
static __device__ __forceinline__ v16h load_a_row(const h16* __restrict__ rowp, int kbase, int lane) {
  int s = (lane >> 4) << 3;
  v8h lo = *(const v8h*)(rowp + kbase + s);
  v8h hi = *(const v8h*)(rowp + kbase + s + 16);
  return mk16(lo, hi);
}
static __device__ __forceinline__ void atom_add_f32(float* p, float v) {
  __hip_atomic_fetch_add(p, v, __ATOMIC_RELAXED, __HIP_MEMORY_SCOPE_AGENT);
}

__global__ void k_cvt(const float* __restrict__ src, h16* __restrict__ dst, int n) {
  int i = blockIdx.x * 256 + threadIdx.x;
  if (i < n) dst[i] = (h16)src[i];
}
__global__ void k_zero(float* __restrict__ p, int n) {
  int i = blockIdx.x * 256 + threadIdx.x;
  if (i < n) p[i] = 0.f;
}

// Encoder: layer1 (tiny K) in VALU -> LDS fp16, layer2 64x64 WMMA -> out0 (+copy out1)
__global__ __launch_bounds__(256) void k_encode(
    const float* __restrict__ X, int din, int M,
    const float* __restrict__ W1, const float* __restrict__ b1,
    const h16* __restrict__ W2, const float* __restrict__ b2,
    h16* __restrict__ out0, h16* __restrict__ out1)
{
  __shared__ h16 lds[8 * 16 * 72];
  int lane = threadIdx.x & 31, wv = threadIdx.x >> 5;
  int base = (blockIdx.x * 8 + wv) * 16;
  h16* L = lds + wv * 16 * 72;

  for (int idx = lane; idx < 16 * 64; idx += 32) {
    int r = idx >> 6, o = idx & 63;
    int rr = base + r; if (rr >= M) rr = M - 1;
    float acc = b1[o];
    for (int j = 0; j < din; ++j) acc += X[(size_t)rr * din + j] * W1[o * din + j];
    L[r * 72 + o] = (h16)(acc > 0.f ? acc : 0.f);
  }
  __syncthreads();

  v8f acc[4] = {};
  const h16* arow = L + (lane & 15) * 72;
  #pragma unroll
  for (int kt = 0; kt < 2; ++kt) {
    v16h a = load_a_row(arow, kt * 32, lane);
    #pragma unroll
    for (int nt = 0; nt < 4; ++nt)
      acc[nt] = wmma32(a, load_b(W2, 64, kt, nt, lane), acc[nt]);
  }
  int hi = lane >> 4, c0 = lane & 15;
  #pragma unroll
  for (int nt = 0; nt < 4; ++nt) {
    int col = nt * 16 + c0;
    float bv = b2[col];
    #pragma unroll
    for (int i = 0; i < 8; ++i) {
      int rowi = base + i + 8 * hi;
      if (rowi < M) {
        float v = acc[nt][i] + bv; v = v > 0.f ? v : 0.f;
        h16 hv = (h16)v;
        out0[(size_t)rowi * 64 + col] = hv;
        if (out1) out1[(size_t)rowi * 64 + col] = hv;
      }
    }
  }
}

// Fused edge update: gather concat(xx[row],xx[col],ee,enc_e) -> MLP(256->64->64) -> ee, scatter recv
__global__ __launch_bounds__(256) void k_edge(
    const int* __restrict__ rowp, const int* __restrict__ colp,
    const h16* __restrict__ xx, const h16* __restrict__ enc_e,
    h16* __restrict__ ee, float* __restrict__ recv,
    const h16* __restrict__ W1, const float* __restrict__ b1,
    const h16* __restrict__ W2, const float* __restrict__ b2)
{
  __shared__ h16 lds[8 * 16 * 72];
  int lane = threadIdx.x & 31, wv = threadIdx.x >> 5;
  int base = (blockIdx.x * 8 + wv) * 16;
  h16* L = lds + wv * 16 * 72;

  int el = base + (lane & 15); if (el >= NE) el = NE - 1;
  int nr = rowp[el], nc = colp[el];
  const h16* seg0 = xx + (size_t)nr * 64;
  const h16* seg1 = xx + (size_t)nc * 64;
  const h16* seg2 = ee + (size_t)el * 64;
  const h16* seg3 = enc_e + (size_t)el * 64;

  v8f acc[4] = {};
  int s8 = (lane >> 4) << 3;
  #pragma unroll
  for (int kt = 0; kt < 8; ++kt) {
    const h16* sp = (kt < 2) ? seg0 : (kt < 4) ? seg1 : (kt < 6) ? seg2 : seg3;
    int off = (kt & 1) * 32 + s8;
    v16h a = mk16(*(const v8h*)(sp + off), *(const v8h*)(sp + off + 16));
    #pragma unroll
    for (int nt = 0; nt < 4; ++nt)
      acc[nt] = wmma32(a, load_b(W1, 256, kt, nt, lane), acc[nt]);
  }
  int hi = lane >> 4, c0 = lane & 15;
  #pragma unroll
  for (int nt = 0; nt < 4; ++nt) {
    int col = nt * 16 + c0;
    float bv = b1[col];
    #pragma unroll
    for (int i = 0; i < 8; ++i) {
      float v = acc[nt][i] + bv;
      L[(i + 8 * hi) * 72 + col] = (h16)(v > 0.f ? v : 0.f);
    }
  }
  __syncthreads();

  v8f a2[4] = {};
  const h16* arow = L + c0 * 72;
  #pragma unroll
  for (int kt = 0; kt < 2; ++kt) {
    v16h a = load_a_row(arow, kt * 32, lane);
    #pragma unroll
    for (int nt = 0; nt < 4; ++nt)
      a2[nt] = wmma32(a, load_b(W2, 64, kt, nt, lane), a2[nt]);
  }
  int cidx[8];
  #pragma unroll
  for (int i = 0; i < 8; ++i) {
    int ei = base + i + 8 * hi;
    cidx[i] = (ei < NE) ? colp[ei] : 0;
  }
  #pragma unroll
  for (int nt = 0; nt < 4; ++nt) {
    int col = nt * 16 + c0;
    float bv = b2[col];
    #pragma unroll
    for (int i = 0; i < 8; ++i) {
      int ei = base + i + 8 * hi;
      if (ei < NE) {
        float v = a2[nt][i] + bv; v = v > 0.f ? v : 0.f;
        ee[(size_t)ei * 64 + col] = (h16)v;
        atom_add_f32(&recv[(size_t)cidx[i] * 64 + col], v);
      }
    }
  }
}

// Fused node update: concat(xx,enc_x,recv) -> MLP(192->64->64) -> xx (in place)
__global__ __launch_bounds__(256) void k_node(
    const h16* __restrict__ enc_x, const float* __restrict__ recv,
    h16* __restrict__ xx,
    const h16* __restrict__ W1, const float* __restrict__ b1,
    const h16* __restrict__ W2, const float* __restrict__ b2)
{
  __shared__ h16 lds[8 * 16 * 72];
  int lane = threadIdx.x & 31, wv = threadIdx.x >> 5;
  int base = (blockIdx.x * 8 + wv) * 16;
  h16* L = lds + wv * 16 * 72;

  int nl = base + (lane & 15); if (nl >= NN) nl = NN - 1;
  const h16* seg0 = xx + (size_t)nl * 64;
  const h16* seg1 = enc_x + (size_t)nl * 64;
  const float* seg2 = recv + (size_t)nl * 64;

  v8f acc[4] = {};
  int s8 = (lane >> 4) << 3;
  #pragma unroll
  for (int kt = 0; kt < 6; ++kt) {
    int off = (kt & 1) * 32 + s8;
    v16h a;
    if (kt < 4) {
      const h16* sp = (kt < 2) ? seg0 : seg1;
      a = mk16(*(const v8h*)(sp + off), *(const v8h*)(sp + off + 16));
    } else {
      v8f f0 = *(const v8f*)(seg2 + off);
      v8f f1 = *(const v8f*)(seg2 + off + 16);
      v8h lo, hi2;
      #pragma unroll
      for (int j = 0; j < 8; ++j) { lo[j] = (h16)f0[j]; hi2[j] = (h16)f1[j]; }
      a = mk16(lo, hi2);
    }
    #pragma unroll
    for (int nt = 0; nt < 4; ++nt)
      acc[nt] = wmma32(a, load_b(W1, 192, kt, nt, lane), acc[nt]);
  }
  int hi = lane >> 4, c0 = lane & 15;
  #pragma unroll
  for (int nt = 0; nt < 4; ++nt) {
    int col = nt * 16 + c0;
    float bv = b1[col];
    #pragma unroll
    for (int i = 0; i < 8; ++i) {
      float v = acc[nt][i] + bv;
      L[(i + 8 * hi) * 72 + col] = (h16)(v > 0.f ? v : 0.f);
    }
  }
  __syncthreads();

  v8f a2[4] = {};
  const h16* arow = L + c0 * 72;
  #pragma unroll
  for (int kt = 0; kt < 2; ++kt) {
    v16h a = load_a_row(arow, kt * 32, lane);
    #pragma unroll
    for (int nt = 0; nt < 4; ++nt)
      a2[nt] = wmma32(a, load_b(W2, 64, kt, nt, lane), a2[nt]);
  }
  #pragma unroll
  for (int nt = 0; nt < 4; ++nt) {
    int col = nt * 16 + c0;
    float bv = b2[col];
    #pragma unroll
    for (int i = 0; i < 8; ++i) {
      int rowi = base + i + 8 * hi;
      if (rowi < NN) {
        float v = a2[nt][i] + bv; v = v > 0.f ? v : 0.f;
        xx[(size_t)rowi * 64 + col] = (h16)v;
      }
    }
  }
}

// Generic K=64 fp16 linear: Y = relu(A @ W^T + b)
__global__ __launch_bounds__(256) void k_lin64(
    const h16* __restrict__ A, const h16* __restrict__ W, const float* __restrict__ b,
    h16* __restrict__ Y, int M)
{
  int lane = threadIdx.x & 31, wv = threadIdx.x >> 5;
  int base = (blockIdx.x * 8 + wv) * 16;
  int rl = base + (lane & 15); if (rl >= M) rl = M - 1;
  const h16* arow = A + (size_t)rl * 64;
  v8f acc[4] = {};
  #pragma unroll
  for (int kt = 0; kt < 2; ++kt) {
    v16h a = load_a_row(arow, kt * 32, lane);
    #pragma unroll
    for (int nt = 0; nt < 4; ++nt)
      acc[nt] = wmma32(a, load_b(W, 64, kt, nt, lane), acc[nt]);
  }
  int hi = lane >> 4, c0 = lane & 15;
  #pragma unroll
  for (int nt = 0; nt < 4; ++nt) {
    int col = nt * 16 + c0;
    float bv = b[col];
    #pragma unroll
    for (int i = 0; i < 8; ++i) {
      int rowi = base + i + 8 * hi;
      if (rowi < M) {
        float v = acc[nt][i] + bv; v = v > 0.f ? v : 0.f;
        Y[(size_t)rowi * 64 + col] = (h16)v;
      }
    }
  }
}

__global__ void k_dec3(const h16* __restrict__ H, const float* __restrict__ W3,
                       const float* __restrict__ b3, float* __restrict__ out, int M) {
  int i = blockIdx.x * 256 + threadIdx.x;
  if (i < M) {
    float acc = b3[0];
    #pragma unroll 8
    for (int j = 0; j < 64; ++j) acc += (float)H[(size_t)i * 64 + j] * W3[j];
    out[i] = acc;
  }
}

extern "C" void kernel_launch(void* const* d_in, const int* in_sizes, int n_in,
                              void* d_out, int out_size, void* d_ws, size_t ws_size,
                              hipStream_t stream) {
  (void)in_sizes; (void)n_in; (void)out_size; (void)ws_size;
  const float* x         = (const float*)d_in[0];
  const float* edge_attr = (const float*)d_in[1];
  const int*   edge_idx  = (const int*)d_in[2];
  const float* ne_W1 = (const float*)d_in[3];
  const float* ne_b1 = (const float*)d_in[4];
  const float* ne_W2 = (const float*)d_in[5];
  const float* ne_b2 = (const float*)d_in[6];
  const float* pe_W1 = (const float*)d_in[7];
  const float* pe_b1 = (const float*)d_in[8];
  const float* pe_W2 = (const float*)d_in[9];
  const float* pe_b2 = (const float*)d_in[10];
  const float* eu_W1 = (const float*)d_in[11];
  const float* eu_b1 = (const float*)d_in[12];
  const float* eu_W2 = (const float*)d_in[13];
  const float* eu_b2 = (const float*)d_in[14];
  const float* nu_W1 = (const float*)d_in[15];
  const float* nu_b1 = (const float*)d_in[16];
  const float* nu_W2 = (const float*)d_in[17];
  const float* nu_b2 = (const float*)d_in[18];
  const float* de_W1 = (const float*)d_in[19];
  const float* de_b1 = (const float*)d_in[20];
  const float* de_W2 = (const float*)d_in[21];
  const float* de_b2 = (const float*)d_in[22];
  const float* de_W3 = (const float*)d_in[23];
  const float* de_b3 = (const float*)d_in[24];

  char* w = (char*)d_ws;
  auto take = [&](size_t bytes) { char* p = w; w += (bytes + 255) & ~(size_t)255; return p; };
  h16* Weu1 = (h16*)take(64 * 256 * 2);
  h16* Weu2 = (h16*)take(64 * 64 * 2);
  h16* Wnu1 = (h16*)take(64 * 192 * 2);
  h16* Wnu2 = (h16*)take(64 * 64 * 2);
  h16* Wpe2 = (h16*)take(64 * 64 * 2);
  h16* Wne2 = (h16*)take(64 * 64 * 2);
  h16* Wd1  = (h16*)take(64 * 64 * 2);
  h16* Wd2  = (h16*)take(64 * 64 * 2);
  h16* enc_x = (h16*)take((size_t)NN * 64 * 2);
  h16* xx    = (h16*)take((size_t)NN * 64 * 2);
  h16* enc_e = (h16*)take((size_t)NE * 64 * 2);
  h16* ee    = (h16*)take((size_t)NE * 64 * 2);
  float* recv = (float*)take((size_t)NN * 64 * 4);
  h16* t1 = (h16*)take((size_t)NN * 64 * 2);
  h16* t2 = (h16*)take((size_t)NN * 64 * 2);

  auto cvt = [&](const float* s, h16* d, int n) {
    k_cvt<<<(n + 255) / 256, 256, 0, stream>>>(s, d, n);
  };
  cvt(eu_W1, Weu1, 64 * 256);
  cvt(eu_W2, Weu2, 64 * 64);
  cvt(nu_W1, Wnu1, 64 * 192);
  cvt(nu_W2, Wnu2, 64 * 64);
  cvt(pe_W2, Wpe2, 64 * 64);
  cvt(ne_W2, Wne2, 64 * 64);
  cvt(de_W1, Wd1, 64 * 64);
  cvt(de_W2, Wd2, 64 * 64);

  int nblkN = ((NN + 15) / 16 + 7) / 8;
  int nblkE = ((NE + 15) / 16 + 7) / 8;

  k_encode<<<nblkN, 256, 0, stream>>>(x, 1, NN, ne_W1, ne_b1, Wne2, ne_b2, enc_x, xx);
  k_encode<<<nblkE, 256, 0, stream>>>(edge_attr, 3, NE, pe_W1, pe_b1, Wpe2, pe_b2, enc_e, ee);

  const int* rowp = edge_idx;
  const int* colp = edge_idx + NE;

  for (int step = 0; step < 7; ++step) {
    k_zero<<<(NN * 64 + 255) / 256, 256, 0, stream>>>(recv, NN * 64);
    k_edge<<<nblkE, 256, 0, stream>>>(rowp, colp, xx, enc_e, ee, recv, Weu1, eu_b1, Weu2, eu_b2);
    k_node<<<nblkN, 256, 0, stream>>>(enc_x, recv, xx, Wnu1, nu_b1, Wnu2, nu_b2);
  }
  k_lin64<<<nblkN, 256, 0, stream>>>(xx, Wd1, de_b1, t1, NN);
  k_lin64<<<nblkN, 256, 0, stream>>>(t1, Wd2, de_b2, t2, NN);
  k_dec3<<<(NN + 255) / 256, 256, 0, stream>>>(t2, de_W3, de_b3, (float*)d_out, NN);
}
